// Engram_6708738917090
// MI455X (gfx1250) — matrix operand order repaired and checked by reading, not compiled, not run on hardware
//
#include <hip/hip_runtime.h>

// ---------------------------------------------------------------------------
// Engram block for MI455X (gfx1250, wave32, WMMA).
// GEMMs run in bf16 via v_wmma_f32_16x16x32_bf16 (f32 accumulate); everything
// else f32. Scratch layout in d_ws (needs ~316 MiB):
//   [0,   8M)   embb      bf16  A matrix (4096 x 1024)
//   [8M,  24M)  keyT      bf16  (G, 2048, 1024)  = key_w transposed per branch
//   [24M, 28M)  valT      bf16  (2048, 1024)     = value_w transposed
//   [28M, 156M) keys_ws   f32   (4096, G*2048)
//   [156M,188M) value_ws  f32   (4096, 2048)
//   [188M,316M) xn_ws     f32   (4096, G*2048)
// ---------------------------------------------------------------------------

typedef __attribute__((ext_vector_type(16))) __bf16 bf16x16;
typedef __attribute__((ext_vector_type(8)))  __bf16 bf16x8;
typedef __attribute__((ext_vector_type(8)))  float  v8f;

#define BSROWS 4096   // B*S
#define GC     8192   // G*C
#define CDIM   2048
#define EDIM   1024   // ENGRAM_DIM (= GEMM K)
#define SLEN   2048

__constant__ int kOffsets[16] = {
    0, 100003, 200022, 300065, 400114, 500171, 600240, 700343,
    800452, 900581, 1000732, 1100885, 1201054, 1301237, 1401426, 1501619};

__device__ __forceinline__ unsigned short f2bf(float f) {
  unsigned u = __builtin_bit_cast(unsigned, f);
  u += 0x7FFFu + ((u >> 16) & 1u);          // round-to-nearest-even
  return (unsigned short)(u >> 16);
}

// ---------------- 1. gather + convert to bf16 -------------------------------
__global__ __launch_bounds__(256) void gather_cvt(
    const int* __restrict__ ids, const float* __restrict__ table,
    unsigned short* __restrict__ embb) {
  const int m = blockIdx.x;            // row in [0, 4096)
  const int t = threadIdx.x;
  const int base = t * 4;              // 4 floats per thread, 1024 per row
  const int h = base >> 6;             // head
  const int off = base & 63;
  const int id = ids[m * 16 + h] + kOffsets[h];
  const float4 v = *(const float4*)(table + (size_t)id * 64 + off);
  ushort4 o;
  o.x = f2bf(v.x); o.y = f2bf(v.y); o.z = f2bf(v.z); o.w = f2bf(v.w);
  *(ushort4*)(embb + (size_t)m * EDIM + base) = o;
}

// ---------------- 2. weight transpose + convert -----------------------------
// src (E=1024, C=2048) f32 row-major  ->  dst (C, E) bf16 row-major.
// z in [0,4): key_w branch g=z; z==4: value_w.
__global__ __launch_bounds__(256) void transpose_cvt(
    const float* __restrict__ key_w, const float* __restrict__ value_w,
    unsigned short* __restrict__ keyT, unsigned short* __restrict__ valT) {
  __shared__ float tile[32][33];
  const int z = blockIdx.z;
  const float* __restrict__ src =
      (z < 4) ? key_w + (size_t)z * EDIM * CDIM : value_w;
  unsigned short* __restrict__ dst =
      (z < 4) ? keyT + (size_t)z * CDIM * EDIM : valT;
  const int o0 = blockIdx.x * 32;
  const int e0 = blockIdx.y * 32;
  const int tx = threadIdx.x, ty = threadIdx.y;  // (32, 8)
#pragma unroll
  for (int i = 0; i < 4; ++i) {
    int e = ty * 4 + i;
    tile[e][tx] = src[(size_t)(e0 + e) * CDIM + o0 + tx];
  }
  __syncthreads();
#pragma unroll
  for (int i = 0; i < 4; ++i) {
    int o = ty * 4 + i;
    dst[(size_t)(o0 + o) * EDIM + e0 + tx] = f2bf(tile[tx][o]);
  }
}

// ---------------- 3. bf16 WMMA GEMM + bias ----------------------------------
// C_out = A(M=4096 x K=1024) * Bt^T + bias, Bt stored (N x K) bf16.
// Block tile 64M x 128N, 8 waves as 2(M) x 4(N); each wave 32x32 (2x2 WMMA
// subtiles).  K staged 64 at a time (double-pumped): one barrier pair per
// 8 wmma / 16 ds_load_b128.  Live VGPR set ~104 (32 acc + 32 frags + 24
// prefetch + addressing) — fits the backend's 128-VGPR budget, no spills.
__global__ __launch_bounds__(256) void gemm_bf16(
    const unsigned short* __restrict__ A,
    const unsigned short* __restrict__ keyT,
    const unsigned short* __restrict__ valT,
    const float* __restrict__ key_b, const float* __restrict__ val_b,
    float* __restrict__ keys_out, float* __restrict__ val_out) {
  constexpr int K   = EDIM;
  constexpr int LDW = 72;  // padded LDS row stride (halfs): 144B, 16B-aligned
  __shared__ __attribute__((aligned(16))) unsigned short Al[64 * LDW];
  __shared__ __attribute__((aligned(16))) unsigned short Bl[128 * LDW];

  const int z = blockIdx.z;
  const unsigned short* __restrict__ Bt =
      (z < 4) ? keyT + (size_t)z * CDIM * K : valT;
  const float* __restrict__ bias = (z < 4) ? key_b + z * CDIM : val_b;

  const int t = threadIdx.x;
  const int lane = t & 31;
  const int w = t >> 5;
  const int wm = w & 1;   // wave M index (0..1)
  const int wn = w >> 1;  // wave N index (0..3)
  const int m0 = blockIdx.y * 64;
  const int n0 = blockIdx.x * 128;

  // cooperative load assignments for a 64-deep K stage
  const int rowA = t >> 2, colA = (t & 3) * 16;  // A: 64 rows x 64 halfs
  const int rowB = t >> 1, colB = (t & 1) * 32;  // B: 128 rows x 64 halfs
  const size_t aBase = (size_t)(m0 + rowA) * K + colA;
  const size_t bBase = (size_t)(n0 + rowB) * K + colB;

  uint4 pa0 = *(const uint4*)(A + aBase);
  uint4 pa1 = *(const uint4*)(A + aBase + 8);
  uint4 pb[4];
#pragma unroll
  for (int j = 0; j < 4; ++j) pb[j] = *(const uint4*)(Bt + bBase + j * 8);

  v8f acc[2][2] = {};

  const int mr  = lane & 15;
  const int kbA = (lane >> 4) * 16;  // A frag: contiguous 16 K-values
  const int kbB = (lane >> 4) * 16;  // B frag: contiguous 16 K-values

  for (int k0 = 0; k0 < K; k0 += 64) {
    __syncthreads();
    *(uint4*)&Al[rowA * LDW + colA]     = pa0;
    *(uint4*)&Al[rowA * LDW + colA + 8] = pa1;
#pragma unroll
    for (int j = 0; j < 4; ++j) *(uint4*)&Bl[rowB * LDW + colB + j * 8] = pb[j];
    __syncthreads();
    if (k0 + 64 < K) {  // prefetch next 64-deep stage, overlapped with WMMAs
      pa0 = *(const uint4*)(A + aBase + k0 + 64);
      pa1 = *(const uint4*)(A + aBase + k0 + 72);
#pragma unroll
      for (int j = 0; j < 4; ++j)
        pb[j] = *(const uint4*)(Bt + bBase + k0 + 64 + j * 8);
    }
#pragma unroll
    for (int kk = 0; kk < 64; kk += 32) {
      bf16x16 a[2], b[2];
#pragma unroll
      for (int mt = 0; mt < 2; ++mt) {
        const unsigned short* p =
            &Al[(wm * 32 + mt * 16 + mr) * LDW + kk + kbA];
        bf16x8 lo = *(const bf16x8*)p;
        bf16x8 hi = *(const bf16x8*)(p + 8);
        a[mt] = __builtin_shufflevector(lo, hi, 0, 1, 2, 3, 4, 5, 6, 7,
                                        8, 9, 10, 11, 12, 13, 14, 15);
      }
#pragma unroll
      for (int nt = 0; nt < 2; ++nt) {
        const unsigned short* p =
            &Bl[(wn * 32 + nt * 16 + mr) * LDW + kk + kbB];
        bf16x8 lo = *(const bf16x8*)p;
        bf16x8 hi = *(const bf16x8*)(p + 8);
        b[nt] = __builtin_shufflevector(lo, hi, 0, 1, 2, 3, 4, 5, 6, 7,
                                        8, 9, 10, 11, 12, 13, 14, 15);
      }
#pragma unroll
      for (int mt = 0; mt < 2; ++mt)
#pragma unroll
        for (int nt = 0; nt < 2; ++nt)
          acc[mt][nt] = __builtin_amdgcn_wmma_f32_16x16x32_bf16(
              false, a[mt], false, b[nt], (short)0, acc[mt][nt], false, false);
    }
  }

  // epilogue: C/D layout — lanes 0-15: N=lane, M=vgpr r; lanes 16-31: M=8+r
  const int nl  = lane & 15;
  const int mhi = (lane >> 4) * 8;
#pragma unroll
  for (int nt = 0; nt < 2; ++nt) {
    const int col = n0 + wn * 32 + nt * 16 + nl;
    const float bv = bias[col];
#pragma unroll
    for (int mt = 0; mt < 2; ++mt) {
      const int r0 = m0 + wm * 32 + mt * 16 + mhi;
#pragma unroll
      for (int r = 0; r < 8; ++r) {
        const float vres = acc[mt][nt][r] + bv;
        if (z < 4)
          keys_out[(size_t)(r0 + r) * GC + z * CDIM + col] = vres;
        else
          val_out[(size_t)(r0 + r) * CDIM + col] = vres;
      }
    }
  }
}

// ---------------- 4. gate + v + conv-input RMSNorm --------------------------
__global__ __launch_bounds__(256) void gate_v(
    const float* __restrict__ hidden, const float* __restrict__ keys,
    const float* __restrict__ value, const float* __restrict__ k_scale,
    const float* __restrict__ q_scale, const float* __restrict__ conv_scale,
    float* __restrict__ out, float* __restrict__ xn) {
  __shared__ float4 red[256];
  __shared__ float res[2];
  const int t = threadIdx.x;
  const int gid = blockIdx.x;          // (b*S+s)*G + g
  const int row = gid >> 2, g = gid & 3;
  const size_t baseG = (size_t)row * GC + g * CDIM;
  const size_t baseV = (size_t)row * CDIM;

  float4 acc = {0.f, 0.f, 0.f, 0.f};   // {sum q^2, sum k^2, sum qk', sum val^2}
#pragma unroll
  for (int i = 0; i < 8; ++i) {
    const int c = i * 256 + t;
    const float q  = hidden[baseG + c];
    const float k  = keys[baseG + c];
    const float vv = value[baseV + c];
    const float qs = q_scale[g * CDIM + c];
    const float ks = k_scale[g * CDIM + c];
    acc.x += q * q;
    acc.y += k * k;
    acc.z += (q * qs) * (k * ks);
    acc.w += vv * vv;
  }
  red[t] = acc;
  __syncthreads();
  for (int s2 = 128; s2 > 0; s2 >>= 1) {
    if (t < s2) {
      float4 a = red[t], b = red[t + s2];
      a.x += b.x; a.y += b.y; a.z += b.z; a.w += b.w;
      red[t] = a;
    }
    __syncthreads();
  }
  if (t == 0) {
    const float4 s = red[0];
    const float rq = rsqrtf(s.x * (1.f / CDIM) + 1e-6f);
    const float rk = rsqrtf(s.y * (1.f / CDIM) + 1e-6f);
    const float qk = s.z * rq * rk * 0.02209708691207961f;  // 1/sqrt(2048)
    const float sgn = (qk > 0.f) ? 1.f : ((qk < 0.f) ? -1.f : 0.f);
    const float gl = sqrtf(fmaxf(fabsf(qk), 1e-6f)) * sgn;
    const float gate = 1.f / (1.f + expf(-gl));
    // mean(v^2) over C == gate^2 * mean(value^2)
    const float invc = rsqrtf(gate * gate * (s.w * (1.f / CDIM)) + 1e-5f);
    res[0] = gate; res[1] = invc;
  }
  __syncthreads();
  const float gate = res[0], invc = res[1];
#pragma unroll
  for (int i = 0; i < 8; ++i) {
    const int c = i * 256 + t;
    const float v = gate * value[baseV + c];
    out[baseG + c] = v;                                   // d_out holds v
    xn[baseG + c] = v * invc * conv_scale[g * CDIM + c];  // RMSNormed conv in
  }
}

// ---------------- 5. causal dilated depthwise conv + SiLU, out = v + y ------
__global__ __launch_bounds__(256) void conv_silu(
    const float* __restrict__ xn, const float* __restrict__ conv_w,
    float* __restrict__ out) {
  const size_t idx4 = (size_t)blockIdx.x * 256 + threadIdx.x;
  const size_t base = idx4 * 4;                 // 4 consecutive gc per thread
  const int row = (int)(base >> 13);            // b*S+s
  const int gc = (int)(base & (GC - 1));
  const int s = row & (SLEN - 1);
  float4 acc = {0.f, 0.f, 0.f, 0.f};
#pragma unroll
  for (int j = 0; j < 4; ++j) {
    const int d = 9 - 3 * j;                    // taps at s-9, s-6, s-3, s
    if (s >= d) {
      const float4 x = *(const float4*)&xn[(size_t)(row - d) * GC + gc];
      const float4 wv = *(const float4*)&conv_w[j * GC + gc];
      acc.x += wv.x * x.x; acc.y += wv.y * x.y;
      acc.z += wv.z * x.z; acc.w += wv.w * x.w;
    }
  }
  float4 o = *(float4*)&out[base];
  o.x += acc.x / (1.f + expf(-acc.x));
  o.y += acc.y / (1.f + expf(-acc.y));
  o.z += acc.z / (1.f + expf(-acc.z));
  o.w += acc.w / (1.f + expf(-acc.w));
  *(float4*)&out[base] = o;
}

// ---------------------------------------------------------------------------
extern "C" void kernel_launch(void* const* d_in, const int* in_sizes, int n_in,
                              void* d_out, int out_size, void* d_ws,
                              size_t ws_size, hipStream_t stream) {
  const float* hidden     = (const float*)d_in[0];
  const int*   ids        = (const int*)d_in[1];
  const float* emb_table  = (const float*)d_in[2];
  const float* key_w      = (const float*)d_in[3];
  const float* key_b      = (const float*)d_in[4];
  const float* k_scale    = (const float*)d_in[5];
  const float* q_scale    = (const float*)d_in[6];
  const float* value_w    = (const float*)d_in[7];
  const float* value_b    = (const float*)d_in[8];
  const float* conv_scale = (const float*)d_in[9];
  const float* conv_w     = (const float*)d_in[10];
  float* out = (float*)d_out;

  char* ws = (char*)d_ws;
  unsigned short* embb = (unsigned short*)ws;                         //   8 MiB
  unsigned short* keyT = (unsigned short*)(ws + (size_t)8 * 1048576); //  16 MiB
  unsigned short* valT = (unsigned short*)(ws + (size_t)24 * 1048576);//   4 MiB
  float* keys_ws  = (float*)(ws + (size_t)28 * 1048576);              // 128 MiB
  float* value_ws = (float*)(ws + (size_t)156 * 1048576);             //  32 MiB
  float* xn_ws    = (float*)(ws + (size_t)188 * 1048576);             // 128 MiB

  gather_cvt<<<BSROWS, 256, 0, stream>>>(ids, emb_table, embb);
  transpose_cvt<<<dim3(CDIM / 32, EDIM / 32, 5), dim3(32, 8), 0, stream>>>(
      key_w, value_w, keyT, valT);
  gemm_bf16<<<dim3(CDIM / 128, BSROWS / 64, 5), 256, 0, stream>>>(
      embb, keyT, valT, key_b, value_b, keys_ws, value_ws);
  gate_v<<<BSROWS * 4, 256, 0, stream>>>(hidden, keys_ws, value_ws, k_scale,
                                         q_scale, conv_scale, out, xn_ws);
  conv_silu<<<(BSROWS * GC) / (256 * 4), 256, 0, stream>>>(xn_ws, conv_w, out);
}
